// ChartSteinSolver_52020643889812
// MI455X (gfx1250) — 8-bit matrix operand order repaired and checked by reading, compiled, not run on hardware
//
#include <hip/hip_runtime.h>
#include <math.h>
#include <stdint.h>


#define NN        1024
#define NTILES    64
#define KCHUNKS   16
#define NSLICES   8
#define PLANE_B   (NN * NN * (int)sizeof(double))
#define SLICE_B   (2 * NSLICES * NN * NN)
#define ITERS     30

#define MW        4
#define NW        2
#define ABUF_B    (NSLICES * MW * 1024)
#define BBUF_B    (NSLICES * NW * 1024)
#define BUF_B     (ABUF_B + BBUF_B)
#define LDS_TOTAL (2 * BUF_B)
#define PF_PER_WAVE 12

typedef int v8i __attribute__((ext_vector_type(8)));

__global__ __launch_bounds__(256) void slice_left_kernel(
    const double* __restrict__ re, const double* __restrict__ im, int stride,
    signed char* __restrict__ S, double* __restrict__ pw)
{
  __shared__ double red[256];
  __shared__ double sh_inv;
  const int row = blockIdx.x;
  const double* pr = re + (size_t)row * NN * stride;
  const double* pi = im + (size_t)row * NN * stride;

  double m = 0.0;
  for (int k = threadIdx.x; k < NN; k += 256) {
    m = fmax(m, fabs(pr[(size_t)k * stride]));
    m = fmax(m, fabs(pi[(size_t)k * stride]));
  }
  red[threadIdx.x] = m;
  __syncthreads();
  for (int off = 128; off > 0; off >>= 1) {
    if (threadIdx.x < (unsigned)off)
      red[threadIdx.x] = fmax(red[threadIdx.x], red[threadIdx.x + off]);
    __syncthreads();
  }
  if (threadIdx.x == 0) {
    double mx = red[0];
    int e = 0;
    if (mx > 0.0) (void)frexp(mx, &e);
    double s = ldexp(2.0, e);
    pw[row] = s;
    sh_inv = 1.0 / s;
  }
  __syncthreads();
  const double inv = sh_inv;

  const int mT = row >> 4, lm = row & 15;
  for (int k = threadIdx.x; k < NN; k += 256) {
    const int kT = k >> 6, kk = k & 63;
    const int b = kk & 3;
    const int v = ((kk >> 2) & 3) | (((kk >> 5) & 1) << 2);
    const int h = (kk >> 4) & 1;
    const size_t tbase = ((size_t)(mT * 16 + kT)) * 1024 + (size_t)(lm + 16 * h) * 32 + v * 4 + b;
    #pragma unroll
    for (int p = 0; p < 2; ++p) {
      double x = (p ? pi : pr)[(size_t)k * stride];
      double r0 = x * inv;
      #pragma unroll
      for (int t = 0; t < NSLICES; ++t) {
        double z = r0 * 128.0;
        double q = rint(z);
        r0 = z - q;
        S[((size_t)(p * NSLICES + t) << 20) + tbase] = (signed char)(int)q;
      }
    }
  }
}

__global__ __launch_bounds__(256) void slice_right_kernel(
    const double* __restrict__ re, const double* __restrict__ im, int stride,
    signed char* __restrict__ S, double* __restrict__ pw)
{
  __shared__ double red[256];
  __shared__ double sh_inv;
  const int col = blockIdx.x;

  double m = 0.0;
  for (int k = threadIdx.x; k < NN; k += 256) {
    size_t idx = ((size_t)k * NN + col) * stride;
    m = fmax(m, fabs(re[idx]));
    m = fmax(m, fabs(im[idx]));
  }
  red[threadIdx.x] = m;
  __syncthreads();
  for (int off = 128; off > 0; off >>= 1) {
    if (threadIdx.x < (unsigned)off)
      red[threadIdx.x] = fmax(red[threadIdx.x], red[threadIdx.x + off]);
    __syncthreads();
  }
  if (threadIdx.x == 0) {
    double mx = red[0];
    int e = 0;
    if (mx > 0.0) (void)frexp(mx, &e);
    double s = ldexp(2.0, e);
    pw[col] = s;
    sh_inv = 1.0 / s;
  }
  __syncthreads();
  const double inv = sh_inv;

  const int nT = col >> 4, ln = col & 15;
  for (int k = threadIdx.x; k < NN; k += 256) {
    const int kT = k >> 6, kk = k & 63;
    const int b = kk & 3;
    const int v = ((kk >> 2) & 3) | (((kk >> 5) & 1) << 2);
    const int h = (kk >> 4) & 1;
    const size_t tbase = ((size_t)(nT * 16 + kT)) * 1024 + (size_t)(ln + 16 * h) * 32 + v * 4 + b;
    size_t idx = ((size_t)k * NN + col) * stride;
    #pragma unroll
    for (int p = 0; p < 2; ++p) {
      double x = (p ? im : re)[idx];
      double r0 = x * inv;
      #pragma unroll
      for (int t = 0; t < NSLICES; ++t) {
        double z = r0 * 128.0;
        double q = rint(z);
        r0 = z - q;
        S[((size_t)(p * NSLICES + t) << 20) + tbase] = (signed char)(int)q;
      }
    }
  }
}

__global__ __launch_bounds__(256) void ozaki_cgemm_kernel(
    const signed char* __restrict__ SL, const signed char* __restrict__ SR,
    const double* __restrict__ powL, const double* __restrict__ powR,
    const double* __restrict__ Rre, const double* __restrict__ Rim, int beta,
    double* __restrict__ Ore, double* __restrict__ Oim)
{
  extern __shared__ char smem[];
  const int tid  = (int)threadIdx.x;
  const int wave = tid >> 5;
  const int lane = tid & 31;
  const int mi   = wave >> 1;
  const int ni   = wave & 1;
  const int mBlk = (int)blockIdx.x >> 5;
  const int nBlk = (int)blockIdx.x & 31;
  const int mT   = mBlk * MW + mi;
  const int nT   = nBlk * NW + ni;
  const int h    = lane >> 4;
  const int ln   = lane & 15;

  const uint32_t ldsBase = (uint32_t)(uintptr_t)smem;

  const double dsc[8] = {0x1p-14, 0x1p-21, 0x1p-28, 0x1p-35,
                         0x1p-42, 0x1p-49, 0x1p-56, 0x1p-63};

  const double sY = powR[nT * 16 + ln];
  double sX[8];
  #pragma unroll
  for (int r = 0; r < 8; ++r) sX[r] = powL[mT * 16 + r + 8 * h];

  double zre[8], zim[8];
  #pragma unroll
  for (int r = 0; r < 8; ++r) { zre[r] = 0.0; zim[r] = 0.0; }

  auto prefetch = [&](int kT, int bufSel, int pA, int pB) {
    #pragma unroll
    for (int j = 0; j < PF_PER_WAVE; ++j) {
      const int i = wave * PF_PER_WAVE + j;
      const signed char* base;
      int plane, t, ldsoff, gT;
      int f = (i < 64) ? i : (i - 64);
      const int half = f & 1;
      f >>= 1;
      if (i < 64) {
        t = f >> 2;  const int ti = f & 3;
        plane = pA;  base = SL;
        ldsoff = (t * MW + ti) * 1024;
        gT = mBlk * MW + ti;
      } else {
        t = f >> 1;  const int ti = f & 1;
        plane = pB;  base = SR;
        ldsoff = ABUF_B + (t * NW + ti) * 1024;
        gT = nBlk * NW + ti;
      }
      const size_t g = ((size_t)(plane * NSLICES + t) << 20)
                     + ((size_t)(gT * 16 + kT) << 10)
                     + (size_t)(half * 512 + lane * 16);
      const uint32_t l = ldsBase + (uint32_t)(bufSel * BUF_B + ldsoff + half * 512 + lane * 16);
      const uint64_t ga = (uint64_t)(uintptr_t)(base + g);
      asm volatile("global_load_async_to_lds_b128 %0, %1, off"
                   :: "v"(l), "v"(ga) : "memory");
    }
  };

  const v8i zero8 = {0, 0, 0, 0, 0, 0, 0, 0};

  #pragma unroll
  for (int pp = 0; pp < 4; ++pp) {
    const int pA = pp >> 1;
    const int pB = pp & 1;

    v8i acc[8];
    #pragma unroll
    for (int c = 0; c < 8; ++c) acc[c] = zero8;

    prefetch(0, 0, pA, pB);

    #pragma unroll 1
    for (int kT = 0; kT < KCHUNKS; ++kT) {
      const int buf = kT & 1;
      if (kT < KCHUNKS - 1) {
        prefetch(kT + 1, buf ^ 1, pA, pB);
        asm volatile("s_wait_asynccnt 12" ::: "memory");
      } else {
        asm volatile("s_wait_asynccnt 0" ::: "memory");
      }
      __syncthreads();

      const v8i* aF = (const v8i*)(smem + buf * BUF_B);
      const v8i* bF = (const v8i*)(smem + buf * BUF_B + ABUF_B);
      v8i a[NSLICES], b[NSLICES];
      #pragma unroll
      for (int t = 0; t < NSLICES; ++t) {
        a[t] = aF[(t * MW + mi) * 32 + lane];
        b[t] = bF[(t * NW + ni) * 32 + lane];
      }
      #pragma unroll
      for (int c = 0; c < 8; ++c) {
        #pragma unroll
        for (int t = 0; t <= c; ++t) {
          acc[c] = __builtin_amdgcn_wmma_i32_16x16x64_iu8(
               true, a[t],  true, b[c - t],
              acc[c],  false,  false);
        }
      }
      __syncthreads();
    }

    #pragma unroll
    for (int r = 0; r < 8; ++r) {
      double s = 0.0;
      #pragma unroll
      for (int c = 0; c < 8; ++c) s = fma((double)acc[c][r], dsc[c], s);
      const double wgt = sX[r] * sY;
      if (pA == pB) zre[r] = fma(s, (pA ? -wgt : wgt), zre[r]);
      else          zim[r] = fma(s, wgt, zim[r]);
    }
  }

  #pragma unroll
  for (int r = 0; r < 8; ++r) {
    const size_t o = (size_t)(mT * 16 + r + 8 * h) * NN + (size_t)(nT * 16 + ln);
    double vr = zre[r], vi = zim[r];
    if (beta) { vr += Rre[o]; vi += Rim[o]; }
    Ore[o] = vr;
    Oim[o] = vi;
  }
}

__global__ __launch_bounds__(256) void pack_out_kernel(
    const double* __restrict__ re, const double* __restrict__ im,
    double* __restrict__ out)
{
  size_t i = (size_t)blockIdx.x * 256 + threadIdx.x;
  if (i < (size_t)NN * NN) {
    out[2 * i]     = re[i];
    out[2 * i + 1] = im[i];
  }
}

extern "C" void kernel_launch(void* const* d_in, const int* in_sizes, int n_in,
                              void* d_out, int out_size, void* d_ws, size_t ws_size,
                              hipStream_t stream)
{
  (void)in_sizes; (void)n_in; (void)out_size; (void)ws_size;

  const double* A = (const double*)d_in[0];
  const double* W = (const double*)d_in[1];
  const double* C = (const double*)d_in[2];
  const double* Y = (const double*)d_in[3];

  char* w = (char*)d_ws;
  auto carve = [&](size_t bytes) -> char* {
    char* p = w;
    w += (bytes + 255) & ~(size_t)255;
    return p;
  };

  signed char* SA = (signed char*)carve(SLICE_B);
  signed char* SW = (signed char*)carve(SLICE_B);
  signed char* S1 = (signed char*)carve(SLICE_B);
  signed char* S2 = (signed char*)carve(SLICE_B);
  double* powA = (double*)carve(NN * sizeof(double));
  double* powW = (double*)carve(NN * sizeof(double));
  double* pow1 = (double*)carve(NN * sizeof(double));
  double* pow2 = (double*)carve(NN * sizeof(double));
  double* Tre  = (double*)carve(PLANE_B);
  double* Tim  = (double*)carve(PLANE_B);
  double* Lre  = (double*)carve(PLANE_B);
  double* Lim  = (double*)carve(PLANE_B);
  double* Rre  = (double*)carve(PLANE_B);
  double* Rim  = (double*)carve(PLANE_B);

  const dim3 gSlice(NN), bSlice(256);
  const dim3 gGemm((NTILES / MW) * (NTILES / NW)), bGemm(256);

  slice_left_kernel<<<gSlice, bSlice, 0, stream>>>(A, A + 1, 2, SA, powA);
  slice_right_kernel<<<gSlice, bSlice, 0, stream>>>(W, W + 1, 2, SW, powW);

  slice_left_kernel<<<gSlice, bSlice, 0, stream>>>(C, C + 1, 2, S1, pow1);
  slice_right_kernel<<<gSlice, bSlice, 0, stream>>>(Y, Y + 1, 2, S2, pow2);
  ozaki_cgemm_kernel<<<gGemm, bGemm, LDS_TOTAL, stream>>>(
      S1, S2, pow1, pow2, Rre, Rim, 0, Lre, Lim);
  hipMemcpyAsync(Rre, Lre, PLANE_B, hipMemcpyDeviceToDevice, stream);
  hipMemcpyAsync(Rim, Lim, PLANE_B, hipMemcpyDeviceToDevice, stream);

  for (int it = 0; it < ITERS; ++it) {
    slice_right_kernel<<<gSlice, bSlice, 0, stream>>>(Lre, Lim, 1, S2, pow2);
    ozaki_cgemm_kernel<<<gGemm, bGemm, LDS_TOTAL, stream>>>(
        SA, S2, powA, pow2, Rre, Rim, 0, Tre, Tim);
    slice_left_kernel<<<gSlice, bSlice, 0, stream>>>(Tre, Tim, 1, S1, pow1);
    ozaki_cgemm_kernel<<<gGemm, bGemm, LDS_TOTAL, stream>>>(
        S1, SW, pow1, powW, Rre, Rim, 1, Lre, Lim);
  }

  pack_out_kernel<<<dim3((NN * NN + 255) / 256), dim3(256), 0, stream>>>(
      Lre, Lim, (double*)d_out);
}
